// LiquidBlock_87076166960156
// MI455X (gfx1250) — compile-verified, matrix operand order checked by reading, not run on hardware
//
#include <hip/hip_runtime.h>

typedef __attribute__((ext_vector_type(16))) __bf16 v16bf;
typedef __attribute__((ext_vector_type(8)))  float  v8f;
typedef __attribute__((ext_vector_type(4)))  unsigned uint4v;
typedef __attribute__((ext_vector_type(2)))  unsigned uint2v;
typedef __attribute__((ext_vector_type(4)))  float  float4v;

#define DEVINL __device__ __forceinline__

namespace {
constexpr int kB = 2, kT = 2048, kH = 1024, kDIN = 2048, kN = 16;
constexpr int kBT = kB * kT;
constexpr int BM = 128, BN = 64, BK = 32;   // block tile; BK == WMMA K for bf16
constexpr int TC = 64;                      // scan time-chunk staged in LDS
}

DEVINL unsigned short f32_to_bf16(float f) {
  unsigned u = __float_as_uint(f);
  u += 0x7FFFu + ((u >> 16) & 1u);          // round-to-nearest-even
  return (unsigned short)(u >> 16);
}
DEVINL float bf16_to_f32(unsigned short h) {
  return __uint_as_float((unsigned)h << 16);
}
DEVINL float sigmoidf_(float x) { return 1.0f / (1.0f + __expf(-x)); }

union AFrag { v16bf v; unsigned u[8]; };
union CAcc  { v8f   v; float    f[8]; };

// ---------------------------------------------------------------- converters
// 8 floats -> 8 bf16 per thread, vectorized (b128 in, b128 out)
__global__ __launch_bounds__(256)
void cvt_f32_bf16_kernel(const float* __restrict__ in,
                         unsigned short* __restrict__ out, int n) {
  const int i = (blockIdx.x * 256 + threadIdx.x) * 8;
  if (i + 8 > n) {
    for (int k = 0; k < 8 && i + k < n; ++k) out[i + k] = f32_to_bf16(in[i + k]);
    return;
  }
  const float4v* in4 = (const float4v*)(in + i);
  float4v a = in4[0], b = in4[1];
  uint4v p;
  p.x = (unsigned)f32_to_bf16(a.x) | ((unsigned)f32_to_bf16(a.y) << 16);
  p.y = (unsigned)f32_to_bf16(a.z) | ((unsigned)f32_to_bf16(a.w) << 16);
  p.z = (unsigned)f32_to_bf16(b.x) | ((unsigned)f32_to_bf16(b.y) << 16);
  p.w = (unsigned)f32_to_bf16(b.z) | ((unsigned)f32_to_bf16(b.w) << 16);
  *(uint4v*)(out + i) = p;
}

__global__ __launch_bounds__(256)
void pack_bc_kernel(const float* __restrict__ Bp, const float* __restrict__ Cp,
                    unsigned short* __restrict__ out) {
  int idx = blockIdx.x * 256 + threadIdx.x;
  if (idx >= kDIN * 32) return;
  int d = idx >> 5, n = idx & 31;
  float v = (n < kN) ? Bp[d * kN + n] : Cp[d * kN + (n - kN)];
  out[idx] = f32_to_bf16(v);
}

// ---------------------------------------------------------------- RMSNorm
__global__ __launch_bounds__(256)
void rmsnorm_kernel(const float* __restrict__ x, const float* __restrict__ w,
                    unsigned short* __restrict__ xn) {
  const int row = blockIdx.x;
  const int tid = threadIdx.x;
  __shared__ float red[256];
  float s = 0.0f;
  for (int h = tid; h < kH; h += 256) {
    float v = x[(size_t)row * kH + h];
    s += v * v;
  }
  red[tid] = s;
  __syncthreads();
  for (int off = 128; off > 0; off >>= 1) {
    if (tid < off) red[tid] += red[tid + off];
    __syncthreads();
  }
  const float rstd = rsqrtf(red[0] / (float)kH + 1e-6f);
  for (int h = tid; h < kH; h += 256)
    xn[(size_t)row * kH + h] = f32_to_bf16(x[(size_t)row * kH + h] * rstd * w[h]);
}

// ---------------------------------------------------------------- WMMA GEMM
// C[M,Nc] = A[M,K] (bf16, row-major) x Bw[K,Nc] (bf16, row-major), f32 acc.
// Double-buffered LDS pipeline: global b128 loads for slab it+1 are in flight
// while WMMA consumes slab it; one workgroup barrier per K-step.
// MODE 0: in_proj  -> silu both halves; cols<DIN -> x_in bf16 (p0), else silu(z) bf16 (p1)
// MODE 1: A_proj   -> sigmoid(acc + bias(p1)) -> a_mod bf16 (p0)
// MODE 2: B/C proj -> f32 B_in (p0) / C_in (p1), Nc = 32
// MODE 3: out_proj -> out (p0) = resid (p1) + acc, f32
template <int MODE>
__global__ __launch_bounds__(256)
void gemm_wmma_kernel(const unsigned short* __restrict__ A,
                      const unsigned short* __restrict__ Bw,
                      int Nc, int K, void* __restrict__ p0, void* __restrict__ p1) {
  __shared__ alignas(16) unsigned short As[2][BM * BK];   // 2 x 8 KB
  __shared__ alignas(16) unsigned short Bs[2][BN * BK];   // 2 x 4 KB, K-transposed

  const int tid  = threadIdx.x;
  const int lane = tid & 31;
  const int wave = tid >> 5;
  const int wm   = wave & 3;   // 4 waves along M (32 rows each)
  const int wn   = wave >> 2;  // 2 waves along N (32 cols each)
  const int lm   = lane & 15;
  const int hi   = lane >> 4;

  const int m0 = blockIdx.y * BM;
  const int n0 = blockIdx.x * BN;

  // CDNA5 16-bit A-fragment K mapping per VGPR-pair p
  int kb[8];
  #pragma unroll
  for (int p = 0; p < 8; ++p)
    kb[p] = 2 * (p & 3) + ((p & 4) ? 16 : 0) + hi * 8;

  CAcc acc[2][2];
  #pragma unroll
  for (int i = 0; i < 2; ++i)
    #pragma unroll
    for (int j = 0; j < 2; ++j)
      #pragma unroll
      for (int r = 0; r < 8; ++r) acc[i][j].f[r] = 0.0f;

  // ---- staging geometry (all static per thread) ----
  // A slab: 128x32 bf16 = 512 uint4; 2 per thread, row-major into LDS.
  const int aIdx0 = tid;                 // uint4 slot 0
  const int aIdx1 = tid + 256;           // uint4 slot 1
  const int ar0 = aIdx0 >> 2, ac0 = (aIdx0 & 3) * 8;
  const int ar1 = aIdx1 >> 2, ac1 = (aIdx1 & 3) * 8;
  // B slab: 32x64 bf16 = 256 uint4; 1 per thread, stored K-transposed.
  const int bk  = tid >> 3;
  const int bn8 = (tid & 7) * 8;

  uint4v aReg0, aReg1, bReg;
  auto loadRegs = [&](int k0) {
    aReg0 = *(const uint4v*)(A + (size_t)(m0 + ar0) * K + k0 + ac0);
    aReg1 = *(const uint4v*)(A + (size_t)(m0 + ar1) * K + k0 + ac1);
    if (n0 + bn8 < Nc)
      bReg = *(const uint4v*)(Bw + (size_t)(k0 + bk) * Nc + n0 + bn8);
    else
      bReg = (uint4v){0u, 0u, 0u, 0u};
  };
  auto storeLds = [&](int buf) {
    uint4v* As4 = (uint4v*)As[buf];
    As4[aIdx0] = aReg0;
    As4[aIdx1] = aReg1;
    unsigned short* bsp = Bs[buf];
    const unsigned short* br = (const unsigned short*)&bReg;
    #pragma unroll
    for (int e = 0; e < 8; ++e)
      bsp[(bn8 + e) * BK + bk] = br[e];
  };

  const int nIter = K / BK;
  loadRegs(0);
  storeLds(0);

  for (int it = 0; it < nIter; ++it) {
    const int buf = it & 1;
    __syncthreads();                       // LDS[buf] ready for all waves
    if (it + 1 < nIter) loadRegs((it + 1) * BK);   // overlap with WMMA below
    if (it + 2 < nIter) {
      __builtin_prefetch(A  + (size_t)(m0 + ar0) * K + (it + 2) * BK + ac0, 0, 1);
      __builtin_prefetch(Bw + (size_t)((it + 2) * BK + bk) * Nc + n0, 0, 1);
    }

    const unsigned* As32r = (const unsigned*)As[buf];
    const unsigned* Bs32r = (const unsigned*)Bs[buf];
    AFrag af[2], bfr[2];
    #pragma unroll
    for (int i = 0; i < 2; ++i) {
      int row = wm * 32 + i * 16 + lm;
      #pragma unroll
      for (int p = 0; p < 8; ++p)
        af[i].u[p] = As32r[row * (BK / 2) + (kb[p] >> 1)];
    }
    #pragma unroll
    for (int j = 0; j < 2; ++j) {
      int col = wn * 32 + j * 16 + lm;
      #pragma unroll
      for (int p = 0; p < 8; ++p)
        bfr[j].u[p] = Bs32r[col * (BK / 2) + (kb[p] >> 1)];
    }

    #pragma unroll
    for (int i = 0; i < 2; ++i)
      #pragma unroll
      for (int j = 0; j < 2; ++j)
        acc[i][j].v = __builtin_amdgcn_wmma_f32_16x16x32_bf16(
            false, af[i].v, false, bfr[j].v, (short)0, acc[i][j].v, false, false);

    if (it + 1 < nIter) storeLds(buf ^ 1); // safe: buf^1 last read in step it-1, before this step's barrier
  }

  // epilogue (C/D layout: row = r + 8*hi, col = lane&15 within 16x16 tile)
  #pragma unroll
  for (int i = 0; i < 2; ++i) {
    #pragma unroll
    for (int j = 0; j < 2; ++j) {
      #pragma unroll
      for (int r = 0; r < 8; ++r) {
        const int grow = m0 + wm * 32 + i * 16 + r + hi * 8;
        const int gcol = n0 + wn * 32 + j * 16 + lm;
        const float val = acc[i][j].f[r];
        if constexpr (MODE == 0) {
          const float v = val * sigmoidf_(val);          // silu
          if (gcol < kDIN)
            ((unsigned short*)p0)[(size_t)grow * kDIN + gcol] = f32_to_bf16(v);
          else
            ((unsigned short*)p1)[(size_t)grow * kDIN + (gcol - kDIN)] = f32_to_bf16(v);
        } else if constexpr (MODE == 1) {
          const float* bias = (const float*)p1;
          ((unsigned short*)p0)[(size_t)grow * kDIN + gcol] =
              f32_to_bf16(sigmoidf_(val + bias[gcol]));
        } else if constexpr (MODE == 2) {
          if (gcol < Nc) {
            if (gcol < kN) ((float*)p0)[(size_t)grow * kN + gcol] = val;
            else           ((float*)p1)[(size_t)grow * kN + (gcol - kN)] = val;
          }
        } else {
          const float* resid = (const float*)p1;
          ((float*)p0)[(size_t)grow * kH + gcol] =
              resid[(size_t)grow * kH + gcol] + val;
        }
      }
    }
  }
}

// ---------------------------------------------------------------- liquid scan
// One thread per (b, d) channel; h[N] state in registers; B/C broadcast via LDS.
__global__ __launch_bounds__(256)
void scan_kernel(const unsigned short* __restrict__ a_mod,
                 const unsigned short* __restrict__ x_in,
                 const unsigned short* __restrict__ gz,
                 const float* __restrict__ Bc, const float* __restrict__ Cc,
                 const float* __restrict__ A_base, const float* __restrict__ Dp,
                 unsigned short* __restrict__ y_out) {
  const int b = blockIdx.x >> 3;                       // DIN/256 = 8 blocks per batch
  const int d = ((blockIdx.x & 7) << 8) + threadIdx.x;
  __shared__ float sB[TC * kN];
  __shared__ float sC[TC * kN];

  float abase[kN];
  #pragma unroll
  for (int n = 0; n < kN; ++n) abase[n] = sigmoidf_(A_base[n]);
  const float Dd = Dp[d];
  float h[kN];
  #pragma unroll
  for (int n = 0; n < kN; ++n) h[n] = 0.0f;

  for (int t0 = 0; t0 < kT; t0 += TC) {
    __syncthreads();
    for (int i = threadIdx.x; i < TC * kN; i += 256) {
      const size_t base = (size_t)(b * kT + t0) * kN + i;
      sB[i] = Bc[base];
      sC[i] = Cc[base];
    }
    __syncthreads();
    for (int tt = 0; tt < TC; ++tt) {
      const size_t row = (size_t)(b * kT + t0 + tt);
      const float a = bf16_to_f32(a_mod[row * kDIN + d]);
      const float x = bf16_to_f32(x_in[row * kDIN + d]);
      float y = 0.0f;
      #pragma unroll
      for (int n = 0; n < kN; ++n) {
        h[n] = __fmaf_rn(a * abase[n], h[n], sB[tt * kN + n] * x);
        y = __fmaf_rn(h[n], sC[tt * kN + n], y);
      }
      const float g = bf16_to_f32(gz[row * kDIN + d]);
      y_out[row * kDIN + d] = f32_to_bf16((y + Dd * x) * g);
    }
  }
}

// ---------------------------------------------------------------- launcher
extern "C" void kernel_launch(void* const* d_in, const int* in_sizes, int n_in,
                              void* d_out, int out_size, void* d_ws, size_t ws_size,
                              hipStream_t stream) {
  (void)in_sizes; (void)n_in; (void)out_size; (void)ws_size;
  const float* x        = (const float*)d_in[0];
  const float* norm_w   = (const float*)d_in[1];
  const float* in_proj  = (const float*)d_in[2];
  const float* A_proj_w = (const float*)d_in[3];
  const float* A_proj_b = (const float*)d_in[4];
  const float* A_base   = (const float*)d_in[5];
  const float* B_proj   = (const float*)d_in[6];
  const float* C_proj   = (const float*)d_in[7];
  const float* Dvec     = (const float*)d_in[8];
  const float* out_proj = (const float*)d_in[9];

  size_t off = 0;
  auto carve = [&](size_t bytes) -> void* {
    void* p = (char*)d_ws + off;
    off += (bytes + 255) & ~(size_t)255;
    return p;
  };
  unsigned short* w_in    = (unsigned short*)carve((size_t)kH * 2 * kDIN * 2);
  unsigned short* w_A     = (unsigned short*)carve((size_t)kDIN * kDIN * 2);
  unsigned short* w_bc    = (unsigned short*)carve((size_t)kDIN * 32 * 2);
  unsigned short* w_out   = (unsigned short*)carve((size_t)kDIN * kH * 2);
  unsigned short* xn      = (unsigned short*)carve((size_t)kBT * kH * 2);
  unsigned short* x_in_bf = (unsigned short*)carve((size_t)kBT * kDIN * 2);
  unsigned short* gz      = (unsigned short*)carve((size_t)kBT * kDIN * 2);
  unsigned short* a_mod   = (unsigned short*)carve((size_t)kBT * kDIN * 2);
  float*          B_in    = (float*)carve((size_t)kBT * kN * 4);
  float*          C_in    = (float*)carve((size_t)kBT * kN * 4);
  unsigned short* y_bf    = (unsigned short*)carve((size_t)kBT * kDIN * 2);

  const dim3 blk(256);
  const int n1 = kH * 2 * kDIN;
  cvt_f32_bf16_kernel<<<(n1 + 2047) / 2048, blk, 0, stream>>>(in_proj, w_in, n1);
  const int n2 = kDIN * kDIN;
  cvt_f32_bf16_kernel<<<(n2 + 2047) / 2048, blk, 0, stream>>>(A_proj_w, w_A, n2);
  const int n3 = kDIN * kH;
  cvt_f32_bf16_kernel<<<(n3 + 2047) / 2048, blk, 0, stream>>>(out_proj, w_out, n3);
  pack_bc_kernel<<<(kDIN * 32 + 255) / 256, blk, 0, stream>>>(B_proj, C_proj, w_bc);

  rmsnorm_kernel<<<kBT, blk, 0, stream>>>(x, norm_w, xn);

  // xz = xn @ in_proj ; silu both halves -> x_in (bf16), silu(z) (bf16)
  gemm_wmma_kernel<0><<<dim3((2 * kDIN) / BN, kBT / BM), blk, 0, stream>>>(
      xn, w_in, 2 * kDIN, kH, (void*)x_in_bf, (void*)gz);
  // a_mod = sigmoid(x_in @ A_proj + b)
  gemm_wmma_kernel<1><<<dim3(kDIN / BN, kBT / BM), blk, 0, stream>>>(
      x_in_bf, w_A, kDIN, kDIN, (void*)a_mod, (void*)A_proj_b);
  // B_in / C_in = x_in @ [B|C]
  gemm_wmma_kernel<2><<<dim3(1, kBT / BM), blk, 0, stream>>>(
      x_in_bf, w_bc, 32, kDIN, (void*)B_in, (void*)C_in);
  // sequential liquid scan + D-skip + silu(z) gating -> y (bf16)
  scan_kernel<<<kB * (kDIN / 256), blk, 0, stream>>>(
      a_mod, x_in_bf, gz, B_in, C_in, A_base, Dvec, y_bf);
  // out = residual + y @ out_proj
  gemm_wmma_kernel<3><<<dim3(kH / BN, kBT / BM), blk, 0, stream>>>(
      y_bf, w_out, kH, kDIN, (void*)d_out, (void*)x);
}
